// Codebook_18468359373526
// MI455X (gfx1250) — matrix operand order repaired and checked by reading, not compiled, not run on hardware
//
#include <hip/hip_runtime.h>

// Problem constants (from reference): x [32,64,64,64] f32, codebook [1024,64] f32
#define KCODES   1024
#define DIM      64
#define BATCH    32
#define HW       4096          // 64*64
#define NTOK     131072        // 32*64*64
#define NELEM    8388608       // NTOK*DIM
#define TOK_WG   256           // tokens per workgroup (32 per wave, 8 waves)
#define CHUNK    256           // codes per LDS chunk
#define BETA     0.25f

typedef __attribute__((ext_vector_type(16))) __bf16 v16bf;
typedef __attribute__((ext_vector_type(8)))  float  v8f;

union BFVec { v16bf v; uint4 q[2]; };

__device__ __forceinline__ unsigned short f2bf(float f) {
  // round-to-nearest-even fp32 -> bf16
  unsigned int u = __float_as_uint(f);
  u += 0x7FFFu + ((u >> 16) & 1u);
  return (unsigned short)(u >> 16);
}

__global__ void zero_loss_kernel(float* loss) { *loss = 0.0f; }

// One wave (32 lanes) per code row: bf16-convert + half squared norm.
__global__ __launch_bounds__(32)
void prep_codes_kernel(const float* __restrict__ lut,
                       unsigned short* __restrict__ cbf,
                       float* __restrict__ hnorm) {
  const int k    = blockIdx.x;      // 0..1023
  const int lane = threadIdx.x;     // 0..31
  float a = lut[k * DIM + lane];
  float b = lut[k * DIM + 32 + lane];
  cbf[k * DIM + lane]      = f2bf(a);
  cbf[k * DIM + 32 + lane] = f2bf(b);
  float s = a * a + b * b;
  #pragma unroll
  for (int off = 16; off > 0; off >>= 1) s += __shfl_xor(s, off, 32);
  if (lane == 0) hnorm[k] = 0.5f * s;
}

__global__ __launch_bounds__(256)
void vq_main_kernel(const float* __restrict__ x,
                    const float* __restrict__ lut,
                    const unsigned short* __restrict__ cbf,
                    const float* __restrict__ hnorm,
                    float* __restrict__ out,
                    float* __restrict__ qout,
                    float* __restrict__ loss) {
  __shared__ __align__(16) unsigned short Alds[TOK_WG * DIM];   // 32 KB  x tile (token-major bf16)
  __shared__ __align__(16) unsigned short Clds[CHUNK * DIM];    // 32 KB  code chunk bf16
  __shared__ float Nlds[CHUNK];                                 //  1 KB  half norms
  __shared__ int   BestIdx[TOK_WG];                             //  1 KB  argmin per token

  const int t    = threadIdx.x;          // 0..255
  const int g    = blockIdx.x;           // 0..511
  const int b    = g >> 4;               // batch image (16 blocks per image)
  const int hw0  = (g & 15) * TOK_WG;    // first hw position of this block
  const int lane = t & 31;
  const int wave = t >> 5;               // 0..7
  const int half = lane >> 4;            // lane half (WMMA operand layout)

  // ---- stage x tile: 64 channels x 256 tokens, coalesced 1KB global rows,
  //      transposed bf16 store (A-operand friendly: Alds[token][channel]) ----
  for (int i = 0; i < 64; ++i) {
    int linear = i * 256 + t;
    int c   = linear >> 8;               // channel 0..63
    int tok = linear & 255;              // token within tile
    float xv = x[(size_t)(b * DIM + c) * HW + hw0 + tok];
    Alds[tok * DIM + c] = f2bf(xv);
  }
  __syncthreads();

  // ---- A operands in registers: wave owns tokens wave*32 .. wave*32+31,
  // as two 16-row WMMA tiles. 16-bit A 16x32 layout: lane m=lane&15;
  // elems 0..7 -> ch kc*32+half*8+0..7, elems 8..15 -> +16 channels. ----
  const int m = lane & 15;
  BFVec a0, a1, a2, a3;
  {
    const unsigned short* r0 = &Alds[(wave * 32 + m) * DIM];
    const unsigned short* r1 = &Alds[(wave * 32 + 16 + m) * DIM];
    a0.q[0] = *(const uint4*)(r0 + 0 * 32 + half * 8);
    a0.q[1] = *(const uint4*)(r0 + 0 * 32 + 16 + half * 8);
    a1.q[0] = *(const uint4*)(r0 + 1 * 32 + half * 8);
    a1.q[1] = *(const uint4*)(r0 + 1 * 32 + 16 + half * 8);
    a2.q[0] = *(const uint4*)(r1 + 0 * 32 + half * 8);
    a2.q[1] = *(const uint4*)(r1 + 0 * 32 + 16 + half * 8);
    a3.q[0] = *(const uint4*)(r1 + 1 * 32 + half * 8);
    a3.q[1] = *(const uint4*)(r1 + 1 * 32 + 16 + half * 8);
  }

  // running argmax of score = x.c - 0.5*||c||^2 (== argmin of d2), per row tile
  float bvA[8], bvB[8];
  int   biA[8], biB[8];
  #pragma unroll
  for (int e = 0; e < 8; ++e) {
    bvA[e] = -3.4e38f; biA[e] = 0;
    bvB[e] = -3.4e38f; biB[e] = 0;
  }

  for (int ch = 0; ch < KCODES / CHUNK; ++ch) {
    __syncthreads();
    // cooperative chunk load: 32KB bf16 codes + 256 norms
    const uint4* src = (const uint4*)(cbf + (size_t)ch * CHUNK * DIM);
    uint4* dst = (uint4*)Clds;
    #pragma unroll
    for (int i = 0; i < 8; ++i) dst[i * 256 + t] = src[i * 256 + t];
    Nlds[t] = hnorm[ch * CHUNK + t];
    // prefetch next chunk while this one is consumed (global_prefetch_b8)
    if (ch + 1 < KCODES / CHUNK)
      __builtin_prefetch(cbf + (size_t)(ch + 1) * CHUNK * DIM + t * 64, 0, 0);
    __syncthreads();

    #pragma unroll 2
    for (int tile = 0; tile < CHUNK / 16; ++tile) {
      // 16-bit B 32x16 layout: lane column n=lane&15; elems 0..7 -> ch
      // kc*32 + half*8 + 0..7, elems 8..15 -> +16 channels (matches A).
      const int n = lane & 15;
      const unsigned short* cb = &Clds[(tile * 16 + n) * DIM + half * 8];
      BFVec b0, b1;
      b0.q[0] = *(const uint4*)(cb);
      b0.q[1] = *(const uint4*)(cb + 16);
      b1.q[0] = *(const uint4*)(cb + 32);
      b1.q[1] = *(const uint4*)(cb + 48);

      // Two independent accumulator chains -> scheduler fills WMMA hazard
      // slots with the other chain instead of v_nops; each B reused 2x.
      v8f accA = {}, accB = {};
      accA = __builtin_amdgcn_wmma_f32_16x16x32_bf16(false, a0.v, false, b0.v,
                                                     (short)0, accA, false, false);
      accB = __builtin_amdgcn_wmma_f32_16x16x32_bf16(false, a2.v, false, b0.v,
                                                     (short)0, accB, false, false);
      accA = __builtin_amdgcn_wmma_f32_16x16x32_bf16(false, a1.v, false, b1.v,
                                                     (short)0, accA, false, false);
      accB = __builtin_amdgcn_wmma_f32_16x16x32_bf16(false, a3.v, false, b1.v,
                                                     (short)0, accB, false, false);

      const int   cand = ch * CHUNK + tile * 16 + n;
      const float hn   = Nlds[tile * 16 + n];
      #pragma unroll
      for (int e = 0; e < 8; ++e) {
        float sA = accA[e] - hn;
        if (sA > bvA[e]) { bvA[e] = sA; biA[e] = cand; }
        float sB = accB[e] - hn;
        if (sB > bvB[e]) { bvB[e] = sB; biB[e] = cand; }
      }
    }
  }

  // ---- cross-lane argmax over the 16 columns in each lane-half.
  // C/D layout: VGPR e -> row e (lanes 0-15) / row 8+e (lanes 16-31). ----
  #pragma unroll
  for (int e = 0; e < 8; ++e) {
    #pragma unroll
    for (int off = 1; off < 16; off <<= 1) {
      float ovA = __shfl_xor(bvA[e], off, 32);
      int   oiA = __shfl_xor(biA[e], off, 32);
      if (ovA > bvA[e] || (ovA == bvA[e] && oiA < biA[e])) { bvA[e] = ovA; biA[e] = oiA; }
      float ovB = __shfl_xor(bvB[e], off, 32);
      int   oiB = __shfl_xor(biB[e], off, 32);
      if (ovB > bvB[e] || (ovB == bvB[e] && oiB < biB[e])) { bvB[e] = ovB; biB[e] = oiB; }
    }
  }
  if ((lane & 15) == 0) {
    #pragma unroll
    for (int e = 0; e < 8; ++e) {
      BestIdx[wave * 32 + half * 8 + e]      = biA[e];
      BestIdx[wave * 32 + 16 + half * 8 + e] = biB[e];
    }
  }
  __syncthreads();

  // q_x (cast to float per output convention): index = b*4096 + hw
  qout[(size_t)b * HW + hw0 + t] = (float)BestIdx[t];

  // ---- gather fp32 code, out = 2x - x_e, loss partials ----
  float ls = 0.0f;
  for (int i = 0; i < 64; ++i) {
    int linear = i * 256 + t;
    int c   = linear >> 8;
    int tok = linear & 255;
    size_t gidx = (size_t)(b * DIM + c) * HW + hw0 + tok;
    float xv = x[gidx];
    float ev = lut[BestIdx[tok] * DIM + c];
    out[gidx] = 2.0f * xv - ev;
    float d = xv - ev;
    ls += d * d;
  }
  #pragma unroll
  for (int off = 16; off > 0; off >>= 1) ls += __shfl_xor(ls, off, 32);
  if (lane == 0)
    atomicAdd(loss, ls * ((1.0f + BETA) / (float)NELEM));
}

extern "C" void kernel_launch(void* const* d_in, const int* in_sizes, int n_in,
                              void* d_out, int out_size, void* d_ws, size_t ws_size,
                              hipStream_t stream) {
  const float* x   = (const float*)d_in[0];   // [32,64,64,64]
  const float* lut = (const float*)d_in[1];   // [1024,64]

  float* out  = (float*)d_out;                // 8388608 floats
  float* qout = out + (size_t)NELEM;          // 131072 floats (indices as float)
  float* loss = qout + (size_t)NTOK;          // 1 float

  unsigned short* cbf   = (unsigned short*)d_ws;                    // 128 KB bf16 codebook
  float*          hnorm = (float*)((char*)d_ws + KCODES * DIM * 2); //   4 KB half norms

  zero_loss_kernel<<<1, 1, 0, stream>>>(loss);
  prep_codes_kernel<<<KCODES, 32, 0, stream>>>(lut, cbf, hnorm);
  vq_main_kernel<<<NTOK / TOK_WG, 256, 0, stream>>>(x, lut, cbf, hnorm,
                                                    out, qout, loss);
}
